// self_KNNLoss_78331613544659
// MI455X (gfx1250) — matrix operand order repaired and checked by reading, not compiled, not run on hardware
//
#include <hip/hip_runtime.h>
#include <hip/hip_bf16.h>
#include <math.h>

// ---------------------------------------------------------------------------
// Problem constants (match reference)
// ---------------------------------------------------------------------------
#define B_N      4096
#define D_DIM    768
#define C_DIM    150
#define C_PAD    160          // pad cluster dim to multiple of 32 for WMMA K loop
#define TOPK     20
#define INV_TEMP 2.0f         // 1 / TEMP, TEMP = 0.5
#define THRESH   0.5f
#define EPS_     1e-8f

typedef __attribute__((ext_vector_type(16))) __bf16 v16bf;
typedef __attribute__((ext_vector_type(8)))  __bf16 v8bf;
typedef __attribute__((ext_vector_type(8)))  float  v8f;

static __device__ __forceinline__ unsigned short f32_to_bf16(float f) {
    unsigned int u = __float_as_uint(f);
    unsigned int r = u + 0x7FFFu + ((u >> 16) & 1u);   // round-to-nearest-even
    return (unsigned short)(r >> 16);
}

// ---------------------------------------------------------------------------
// Kernel: zero the scalar output (harness poisons d_out)
// ---------------------------------------------------------------------------
__global__ void zero_out(float* o) {
    if (threadIdx.x == 0) o[0] = 0.0f;
}

// ---------------------------------------------------------------------------
// Kernel: z (fp32 [B,D]) -> bf16 copy + row L2 norms. One wave per row.
// ---------------------------------------------------------------------------
__global__ __launch_bounds__(256)
void prep_z(const float* __restrict__ z, unsigned short* __restrict__ zb,
            float* __restrict__ nrm) {
    const int wave = threadIdx.x >> 5;
    const int lane = threadIdx.x & 31;
    const int row  = blockIdx.x * 8 + wave;
    const float* zr = z + (size_t)row * D_DIM;
    unsigned short* zbr = zb + (size_t)row * D_DIM;
    float s = 0.0f;
    for (int k = lane; k < D_DIM; k += 32) {
        float v = zr[k];
        s += v * v;
        zbr[k] = f32_to_bf16(v);
    }
    #pragma unroll
    for (int off = 16; off; off >>= 1) s += __shfl_xor(s, off, 32);
    if (lane == 0) nrm[row] = sqrtf(s);
}

// ---------------------------------------------------------------------------
// Kernel: c_i (fp32 [B,150]) -> bf16 [B,160] zero-padded. One wave per row.
// ---------------------------------------------------------------------------
__global__ __launch_bounds__(256)
void prep_c(const float* __restrict__ c, unsigned short* __restrict__ cb) {
    const int wave = threadIdx.x >> 5;
    const int lane = threadIdx.x & 31;
    const int row  = blockIdx.x * 8 + wave;
    for (int k = lane; k < C_PAD; k += 32) {
        float v = (k < C_DIM) ? c[(size_t)row * C_DIM + k] : 0.0f;
        cb[(size_t)row * C_PAD + k] = f32_to_bf16(v);
    }
}

// ---------------------------------------------------------------------------
// Gram kernel: out[m,n] = f( A[m,:] . Bm[n,:] )   (A,Bm bf16 row-major, K=KDIM)
// MODE 0: out = exp(dot / max(na[m]*nb[n], eps) * (1/T))
// MODE 1: out = (dot > 0.5 || m==n) ? 1 : 0        (pseudo-label mask)
// Workgroup = 256 threads (8 waves). Tile: 128 (M) x 64 (N).
// Each wave owns a 16-row strip and 4 accumulators (4 x 16x16 tiles).
// ---------------------------------------------------------------------------
template <int KDIM, int MODE>
__global__ __launch_bounds__(256)
void gram_kernel(const unsigned short* __restrict__ A,
                 const unsigned short* __restrict__ Bm,
                 const float* __restrict__ na, const float* __restrict__ nb,
                 float* __restrict__ out) {
    __shared__ __align__(16) unsigned short As[128 * 32];
    __shared__ __align__(16) unsigned short Bs[64 * 32];

    const int tid   = threadIdx.x;
    const int wave  = tid >> 5;
    const int lane  = tid & 31;
    const int half_ = lane >> 4;     // 0: lanes 0-15, 1: lanes 16-31
    const int lr    = lane & 15;

    const int m0 = blockIdx.y * 128;
    const int n0 = blockIdx.x * 64;

    v8f acc[4] = {};

    for (int k0 = 0; k0 < KDIM; k0 += 32) {
        __syncthreads();
        // ---- stage A tile (128x32 bf16 = 8KB): 512 x 16B chunks, 2/thread
        #pragma unroll
        for (int cc = 0; cc < 2; ++cc) {
            int chunk = tid + cc * 256;
            int row   = chunk >> 2;
            int off   = (chunk & 3) * 8;
            const float4* src = (const float4*)(A + (size_t)(m0 + row) * KDIM + k0 + off);
            *(float4*)(&As[row * 32 + off]) = *src;
        }
        // ---- stage B tile (64x32 bf16 = 4KB): 256 x 16B chunks, 1/thread
        {
            int row = tid >> 2;
            int off = (tid & 3) * 8;
            const float4* src = (const float4*)(Bm + (size_t)(n0 + row) * KDIM + k0 + off);
            *(float4*)(&Bs[row * 32 + off]) = *src;
        }
        // ---- prefetch hint for next K tile (global_prefetch_b8)
        if (k0 + 32 < KDIM) {
            __builtin_prefetch(A  + (size_t)(m0 + (tid >> 2)) * KDIM + (k0 + 32), 0, 1);
            __builtin_prefetch(Bm + (size_t)(n0 + (tid >> 2)) * KDIM + (k0 + 32), 0, 1);
        }
        __syncthreads();

        // ---- A fragment (16x32): lane<16 holds K{0..7,16..23}, lane>=16 K{8..15,24..31}
        const int ar = wave * 16 + lr;
        v8bf a_lo = *(const v8bf*)(&As[ar * 32 + 8 * half_]);
        v8bf a_hi = *(const v8bf*)(&As[ar * 32 + 16 + 8 * half_]);
        v16bf afrag = __builtin_shufflevector(a_lo, a_hi,
                          0,1,2,3,4,5,6,7,8,9,10,11,12,13,14,15);

        // ---- 4 B fragments (32x16 each): lane<16 holds K{0..7,16..23}, lane>=16 K{8..15,24..31}, N = lr
        #pragma unroll
        for (int j = 0; j < 4; ++j) {
            const int br = j * 16 + lr;
            v8bf b_lo = *(const v8bf*)(&Bs[br * 32 + 8 * half_]);
            v8bf b_hi = *(const v8bf*)(&Bs[br * 32 + 16 + 8 * half_]);
            v16bf bfrag = __builtin_shufflevector(b_lo, b_hi,
                              0,1,2,3,4,5,6,7,8,9,10,11,12,13,14,15);
            acc[j] = __builtin_amdgcn_wmma_f32_16x16x32_bf16(
                         false, afrag, false, bfrag, (short)0, acc[j], false, false);
        }
    }

    // ---- epilogue: C/D layout -> VGPR i: M = 8*half_ + i, N = lr
    float nam[8];
    if (MODE == 0) {
        #pragma unroll
        for (int i = 0; i < 8; ++i)
            nam[i] = na[m0 + wave * 16 + half_ * 8 + i];
    }
    #pragma unroll
    for (int j = 0; j < 4; ++j) {
        const int gn = n0 + j * 16 + lr;
        const float nbn = (MODE == 0) ? nb[gn] : 0.0f;
        #pragma unroll
        for (int i = 0; i < 8; ++i) {
            const int gm = m0 + wave * 16 + half_ * 8 + i;
            const float dot = acc[j][i];
            float val;
            if (MODE == 0) {
                float denom = fmaxf(nam[i] * nbn, EPS_);
                val = __expf(dot * INV_TEMP / denom);
            } else {
                val = (dot > THRESH || gm == gn) ? 1.0f : 0.0f;
            }
            out[(size_t)gm * B_N + gn] = val;
        }
    }
}

// ---------------------------------------------------------------------------
// Top-(K+1) per row of x. Row cached in LDS, 21 argmax-extract passes.
// The diagonal (cos=1 -> e^2, the row max) is always selected, matching ref.
// ---------------------------------------------------------------------------
__global__ __launch_bounds__(256)
void topk_kernel(const float* __restrict__ x, int* __restrict__ selc) {
    __shared__ float rowv[B_N];
    __shared__ float rmax[256];
    __shared__ int   ridx[256];
    const int row = blockIdx.x;
    const float* xr = x + (size_t)row * B_N;
    for (int k = threadIdx.x; k < B_N; k += 256) rowv[k] = xr[k];
    __syncthreads();

    for (int it = 0; it < TOPK + 1; ++it) {
        float mv = -3.4e38f;
        int   mi = 0x7FFFFFFF;
        for (int k = threadIdx.x; k < B_N; k += 256) {
            float v = rowv[k];
            if (v > mv) { mv = v; mi = k; }
        }
        rmax[threadIdx.x] = mv;
        ridx[threadIdx.x] = mi;
        __syncthreads();
        for (int s = 128; s > 0; s >>= 1) {
            if (threadIdx.x < s) {
                float ov = rmax[threadIdx.x + s];
                int   oi = ridx[threadIdx.x + s];
                if (ov > rmax[threadIdx.x] ||
                    (ov == rmax[threadIdx.x] && oi < ridx[threadIdx.x])) {
                    rmax[threadIdx.x] = ov;
                    ridx[threadIdx.x] = oi;
                }
            }
            __syncthreads();
        }
        if (threadIdx.x == 0) {
            int mi0 = ridx[0];
            selc[row * (TOPK + 1) + it] = mi0;
            rowv[mi0] = -3.4e38f;
        }
        __syncthreads();
    }
}

// ---------------------------------------------------------------------------
// Final loss: one wave per row. Lanes 0..20 own one selected column each.
// x1 = [x[row,sel] (20, non-diag), x_ci[row,sel_c] (21)], x2 analogous with
// x_adv and x_ci^T; m1 from mask (diag already forced to 1 in mask buffer).
// Sums are order-independent, so no column sort is needed.
// ---------------------------------------------------------------------------
__global__ __launch_bounds__(256)
void loss_kernel(const float* __restrict__ x,    const float* __restrict__ xadv,
                 const float* __restrict__ xci,  const float* __restrict__ maskf,
                 const int* __restrict__ selc,   float* __restrict__ out) {
    const int wave = threadIdx.x >> 5;
    const int lane = threadIdx.x & 31;
    const int row  = blockIdx.x * 8 + wave;
    const int K1   = TOPK + 1;

    float xa = 0.f, x2a = 0.f, xb = 0.f, x2b = 0.f, ma = 0.f, mb = 0.f;
    bool active = (lane < K1);
    bool isdiag = false;
    if (active) {
        const int c = selc[row * K1 + lane];
        isdiag = (c == row);
        const size_t ro = (size_t)row * B_N;
        xb  = xci[ro + c];
        x2b = xci[(size_t)c * B_N + row];
        mb  = maskf[ro + c];           // diagonal already 1
        if (!isdiag) {
            xa  = x[ro + c];
            x2a = xadv[ro + c];
            ma  = mb;
        }
    }
    float s1 = xa + xb;        // x1 row sum contributions (41 entries total)
    float s2 = x2a + x2b;
    float mx = ma;             // mask_x sum (20 non-diag part-A entries)
    #pragma unroll
    for (int off = 16; off; off >>= 1) {
        s1 += __shfl_xor(s1, off, 32);
        s2 += __shfl_xor(s2, off, 32);
        mx += __shfl_xor(mx, off, 32);
    }
    float l = 0.0f;
    if (active) {
        float d1 = xb * mb / s1 + (1.0f - mb);
        float d2 = x2b * mb / s2 + (1.0f - mb);
        l = __logf(d1) + __logf(d2);
        if (!isdiag) {
            float e1 = xa * ma / s1 + (1.0f - ma);
            float e2 = x2a * ma / s2 + (1.0f - ma);
            l += __logf(e1) + __logf(e2);
        }
    }
    #pragma unroll
    for (int off = 16; off; off >>= 1) l += __shfl_xor(l, off, 32);
    if (lane == 0) {
        float lrow = l / (mx * 2.0f + 1.0f);
        atomicAdd(out, -lrow * (1.0f / (float)B_N));   // -mean(loss_row)
    }
}

// ---------------------------------------------------------------------------
// Launch
// ---------------------------------------------------------------------------
extern "C" void kernel_launch(void* const* d_in, const int* in_sizes, int n_in,
                              void* d_out, int out_size, void* d_ws, size_t ws_size,
                              hipStream_t stream) {
    const float* z_i = (const float*)d_in[0];   // [4096,768]
    const float* z_j = (const float*)d_in[1];   // [4096,768]
    const float* c_i = (const float*)d_in[2];   // [4096,150]
    (void)d_in; (void)in_sizes; (void)n_in; (void)out_size; (void)ws_size;

    // Workspace carve-out (256B aligned blocks), total ~283 MB.
    char* ws = (char*)d_ws;
    size_t off = 0;
    auto carve = [&](size_t bytes) -> void* {
        void* p = ws + off;
        off = (off + bytes + 255) & ~(size_t)255;
        return p;
    };
    unsigned short* zi_bf = (unsigned short*)carve((size_t)B_N * D_DIM * 2);
    unsigned short* zj_bf = (unsigned short*)carve((size_t)B_N * D_DIM * 2);
    unsigned short* ci_bf = (unsigned short*)carve((size_t)B_N * C_PAD * 2);
    float* ni    = (float*)carve((size_t)B_N * 4);
    float* nj    = (float*)carve((size_t)B_N * 4);
    int*   selc  = (int*)carve((size_t)B_N * (TOPK + 1) * 4);
    float* x     = (float*)carve((size_t)B_N * B_N * 4);
    float* xadv  = (float*)carve((size_t)B_N * B_N * 4);
    float* xci   = (float*)carve((size_t)B_N * B_N * 4);
    float* maskf = (float*)carve((size_t)B_N * B_N * 4);

    float* outp = (float*)d_out;

    zero_out<<<1, 32, 0, stream>>>(outp);

    prep_z<<<B_N / 8, 256, 0, stream>>>(z_i, zi_bf, ni);
    prep_z<<<B_N / 8, 256, 0, stream>>>(z_j, zj_bf, nj);
    prep_c<<<B_N / 8, 256, 0, stream>>>(c_i, ci_bf);

    dim3 gg(B_N / 64, B_N / 128);
    gram_kernel<D_DIM, 0><<<gg, 256, 0, stream>>>(zi_bf, zi_bf, ni, ni, x);
    gram_kernel<D_DIM, 0><<<gg, 256, 0, stream>>>(zj_bf, zj_bf, nj, nj, xadv);
    gram_kernel<D_DIM, 0><<<gg, 256, 0, stream>>>(zi_bf, zj_bf, ni, nj, xci);
    gram_kernel<C_PAD, 1><<<gg, 256, 0, stream>>>(ci_bf, ci_bf, nullptr, nullptr, maskf);

    topk_kernel<<<B_N, 256, 0, stream>>>(x, selc);
    loss_kernel<<<B_N / 8, 256, 0, stream>>>(x, xadv, xci, maskf, selc, outp);
}
